// VEMamba_20040317403817
// MI455X (gfx1250) — matrix operand order repaired and checked by reading, not compiled, not run on hardware
//
#include <hip/hip_runtime.h>
#include <math.h>

// ---------------------------------------------------------------------------
// VEMamba forward for MI455X (gfx1250, wave32).  All GEMMs lowered to
// v_wmma_f32_16x16x32_f16 (f16 in, f32 accumulate), weights staged in LDS.
// ---------------------------------------------------------------------------

typedef __attribute__((ext_vector_type(16))) _Float16 v16h;
typedef __attribute__((ext_vector_type(8)))  float    v8f;

#define Fc   4
#define Hc   32
#define Wc   128
#define DIMc 32
#define DSc  16
#define DIc  64
#define CKc  4
#define Lc   (Fc * Hc * Wc)        // 16384 tokens
#define H2c  (2 * Hc)
#define H4c  (4 * Hc)

static __device__ __forceinline__ float silu_f(float x) {
  return x / (1.0f + __expf(-x));
}

// ---------------------------------------------------------------------------
// WMMA GEMM: C[M,N] = act(A[M,K] @ W[K,N] + bias) + res_scale * res
//   KT = K/32 chunks, NT = ceil(N/16) tiles.  W^T is staged zero-padded in
//   LDS as f16, so B-fragment loads are branch-free aligned ds_load_b128s.
//   One wave per 16-row m-tile; A fragments cached in registers and reused
//   across all NT n-tiles.  Grid is exact: M/16 waves total, 8 waves/block.
//   act: 0 = none, 1 = silu, 2 = exact gelu
// ---------------------------------------------------------------------------
template <int KT, int NT>
__global__ void gemm16_wmma(const float* __restrict__ A,
                            const float* __restrict__ W,
                            const float* __restrict__ bias,
                            const float* __restrict__ res,
                            const float* __restrict__ res_scale,
                            float* __restrict__ C,
                            int M, int N, int act) {
  constexpr int K  = KT * 32;
  constexpr int NP = NT * 16;
  __shared__ _Float16 wlds[NP * K];            // [n][k], n zero-padded to NP

  for (int i = threadIdx.x; i < NP * K; i += blockDim.x) {
    int n = i / K, k = i % K;
    wlds[i] = (n < N) ? (_Float16)W[(size_t)k * N + n] : (_Float16)0.f;
  }
  __syncthreads();

  int wave = threadIdx.x >> 5;
  int lane = threadIdx.x & 31;
  int half = lane >> 4;                        // 0: lanes 0-15, 1: lanes 16-31
  int l15  = lane & 15;
  int tm   = blockIdx.x * (blockDim.x >> 5) + wave;
  int am   = tm * 16 + l15;                    // A-fragment row

  // A fragments (16x32 f16 each): lane half selects kbase 0/16; contiguous 16 k
  v16h af[KT];
  const float* Arow = A + (size_t)am * K;
#pragma unroll
  for (int kt = 0; kt < KT; ++kt) {
    int kbase = kt * 32 + half * 16;
#pragma unroll
    for (int e = 0; e < 16; ++e)
      af[kt][e] = (_Float16)Arow[kbase + e];
  }

#pragma unroll
  for (int tn = 0; tn < NT; ++tn) {
    v8f acc = {0.f, 0.f, 0.f, 0.f, 0.f, 0.f, 0.f, 0.f};
#pragma unroll
    for (int kt = 0; kt < KT; ++kt) {
      // B fragment: lane (n = tn*16+l15) reads K-run [kt*32 + half*16, +16)
      const _Float16* bp = wlds + (size_t)(tn * 16 + l15) * K + kt * 32 + half * 16;
      v16h bf = *(const v16h*)bp;              // 32B aligned contiguous LDS read
      acc = __builtin_amdgcn_wmma_f32_16x16x32_f16(
          false, af[kt], false, bf, (short)0, acc, false, false);
    }
    int bn = tn * 16 + l15;
    if (bn < N) {
      float b  = bias ? bias[bn] : 0.f;
      float sc = res_scale ? res_scale[bn] : 1.f;
#pragma unroll
      for (int v = 0; v < 8; ++v) {
        int m = tm * 16 + v + half * 8;        // C/D layout: M = vgpr + 8*half
        float xv = acc[v] + b;
        if (act == 1)      xv = silu_f(xv);
        else if (act == 2) xv = 0.5f * xv * (1.f + erff(xv * 0.70710678118f));
        float r = res ? res[(size_t)m * N + bn] * sc : 0.f;
        C[(size_t)m * N + bn] = xv + r;
      }
    }
  }
}

// ---------------------------------------------------------------------------
// Channel-last 3D convolution, weights (O,I,KD,KH,KW), optional residual add.
// ---------------------------------------------------------------------------
__global__ void conv3d_cl(const float* __restrict__ in,
                          const float* __restrict__ wgt,
                          const float* __restrict__ bias,
                          const float* __restrict__ res,
                          float* __restrict__ out,
                          int D, int H, int W, int Ci, int Co,
                          int KD, int KH, int KW, int PD, int PH, int PW) {
  int idx = blockIdx.x * blockDim.x + threadIdx.x;
  int total = D * H * W * Co;
  if (idx >= total) return;
  int co = idx % Co;
  int t  = idx / Co;
  int w  = t % W;  t /= W;
  int h  = t % H;
  int d  = t / H;
  float acc = bias ? bias[co] : 0.f;
  int kstride = KD * KH * KW;
  for (int kd = 0; kd < KD; ++kd) {
    int dd = d + kd - PD; if (dd < 0 || dd >= D) continue;
    for (int kh = 0; kh < KH; ++kh) {
      int hh = h + kh - PH; if (hh < 0 || hh >= H) continue;
      for (int kw2 = 0; kw2 < KW; ++kw2) {
        int ww = w + kw2 - PW; if (ww < 0 || ww >= W) continue;
        const float* ip = in + ((size_t)(dd * H + hh) * W + ww) * Ci;
        const float* wp = wgt + (size_t)co * Ci * kstride
                               + (size_t)(kd * KH + kh) * KW + kw2;
        for (int ci = 0; ci < Ci; ++ci)
          acc += ip[ci] * wp[(size_t)ci * kstride];
      }
    }
  }
  out[idx] = acc + (res ? res[idx] : 0.f);
}

// FiLM modulation vector: mod[j] = sum_i cdp[i] * kw[i,j]   (j < Out=2C)
__global__ void mod_k(const float* __restrict__ cdp, const float* __restrict__ kw,
                      float* __restrict__ mod, int In, int Out) {
  int j = blockIdx.x * blockDim.x + threadIdx.x;
  if (j >= Out) return;
  float a = 0.f;
  for (int i = 0; i < In; ++i) a += cdp[i] * kw[(size_t)i * Out + j];
  mod[j] = a;
}

// LayerNorm + FiLM per token (C = 32 channels)
__global__ void vdim_k(const float* __restrict__ t, const float* __restrict__ g,
                       const float* __restrict__ b, const float* __restrict__ mod,
                       float* __restrict__ out, int L, int C) {
  int l = blockIdx.x * blockDim.x + threadIdx.x;
  if (l >= L) return;
  const float* x = t + (size_t)l * C;
  float m = 0.f;
  for (int c = 0; c < C; ++c) m += x[c];
  m /= (float)C;
  float v = 0.f;
  for (int c = 0; c < C; ++c) { float d = x[c] - m; v += d * d; }
  v /= (float)C;
  float inv = rsqrtf(v + 1e-6f);
  float* o = out + (size_t)l * C;
  for (int c = 0; c < C; ++c) {
    float xn = (x[c] - m) * inv * g[c] + b[c];
    o[c] = xn * mod[c] + mod[C + c];
  }
}

// Causal depthwise conv1d over sequence + bias + SiLU; xi from xz[:, :64]
__global__ void dwconv_silu_k(const float* __restrict__ xz,
                              const float* __restrict__ cw,
                              const float* __restrict__ cb,
                              float* __restrict__ xi, int L) {
  int idx = blockIdx.x * blockDim.x + threadIdx.x;
  if (idx >= L * DIc) return;
  int c = idx % DIc;
  int l = idx / DIc;
  float a = 0.f;
#pragma unroll
  for (int k = 0; k < CKc; ++k) {
    int ll = l - (CKc - 1) + k;
    if (ll >= 0) a += xz[(size_t)ll * (2 * DIc) + c] * cw[c * CKc + k];
  }
  a += cb[c];
  xi[idx] = silu_f(a);
}

// delta[l,di] = softplus(dbl[l,0:2] @ dt_w + dt_b)
__global__ void delta_k(const float* __restrict__ dbl, const float* __restrict__ dtw,
                        const float* __restrict__ dtb, float* __restrict__ delta,
                        int L) {
  int idx = blockIdx.x * blockDim.x + threadIdx.x;
  if (idx >= L * DIc) return;
  int c = idx % DIc;
  int l = idx / DIc;
  const float* r = dbl + (size_t)l * 34;
  float a = r[0] * dtw[c] + r[1] * dtw[DIc + c] + dtb[c];
  delta[idx] = (a > 20.f) ? a : log1pf(__expf(a));
}

// Selective scan: 1024 independent (di,s) recurrences; 16-lane DS reduction
// per step.  Block = 256 threads = 16 di x 16 s; grid = DIc/16 blocks.
__global__ void scan_k(const float* __restrict__ delta,
                       const float* __restrict__ dbl,
                       const float* __restrict__ xi,
                       const float* __restrict__ A_log,
                       float* __restrict__ ysc, int L) {
  int tid = threadIdx.x;
  int s   = tid & 15;
  int di  = blockIdx.x * 16 + (tid >> 4);
  float Anat = -__expf(A_log[di * DSc + s]);
  float h = 0.f;
  for (int l = 0; l < L; ++l) {
    float dl  = delta[(size_t)l * DIc + di];
    const float* r = dbl + (size_t)l * 34;
    float Bm  = r[2 + s];
    float Cm  = r[2 + DSc + s];
    float xiv = xi[(size_t)l * DIc + di];
    h = __expf(dl * Anat) * h + dl * Bm * xiv;
    float p = h * Cm;
#pragma unroll
    for (int off = 1; off < 16; off <<= 1)
      p += __shfl_xor(p, off, 32);          // stays within aligned 16-lane group
    if (s == 0) ysc[(size_t)l * DIc + di] = p;
  }
}

// y = (ysc + D*xi) * silu(z),  z = xz[:, 64:]
__global__ void gate_k(const float* __restrict__ ysc, const float* __restrict__ xi,
                       const float* __restrict__ xz, const float* __restrict__ Dp,
                       float* __restrict__ y, int L) {
  int idx = blockIdx.x * blockDim.x + threadIdx.x;
  if (idx >= L * DIc) return;
  int c = idx % DIc;
  int l = idx / DIc;
  float z = xz[(size_t)l * (2 * DIc) + DIc + c];
  y[idx] = (ysc[idx] + Dp[c] * xi[idx]) * silu_f(z);
}

__global__ void add_k(const float* __restrict__ a, const float* __restrict__ b,
                      float* __restrict__ o, int n) {
  int i = blockIdx.x * blockDim.x + threadIdx.x;
  if (i < n) o[i] = a[i] + b[i];
}

// anisotropic pixel shuffle (H x2 from channels) + leaky(0.1)
// in: (D,H,W,2*Co) channel-last  ->  out: (D,2H,W,Co)
__global__ void ps_leaky_k(const float* __restrict__ in, float* __restrict__ out,
                           int D, int H, int W, int Co) {
  int idx = blockIdx.x * blockDim.x + threadIdx.x;
  int total = D * 2 * H * W * Co;
  if (idx >= total) return;
  int co = idx % Co;
  int t  = idx / Co;
  int w  = t % W;  t /= W;
  int h2 = t % (2 * H);
  int d  = t / (2 * H);
  int h = h2 >> 1, rr = h2 & 1;
  float v = in[((size_t)(d * H + h) * W + w) * (2 * Co) + co * 2 + rr];
  out[idx] = (v >= 0.f) ? v : 0.1f * v;
}

// ---------------------------------------------------------------------------
// Host orchestration
// ---------------------------------------------------------------------------
static inline int cdiv(int a, int b) { return (a + b - 1) / b; }

template <int KT, int NT>
static void gemm(const float* A, const float* W, const float* bias,
                 const float* res, const float* rs, float* C,
                 int M, int N, int act, hipStream_t s) {
  // 8 waves (8 m-tiles) per 256-thread block; M = 16384 -> exact grid
  int grid = (M / 16) / 8;
  gemm16_wmma<KT, NT><<<grid, 256, 0, s>>>(A, W, bias, res, rs, C, M, N, act);
}

extern "C" void kernel_launch(void* const* d_in, const int* in_sizes, int n_in,
                              void* d_out, int out_size, void* d_ws, size_t ws_size,
                              hipStream_t stream) {
  (void)in_sizes; (void)n_in; (void)out_size; (void)ws_size;
  auto P = [&](int i) { return (const float*)d_in[i]; };

  // pytree (insertion-order DFS) input indices:
  // 0:x 1:cdp 2:first_w 3:first_b
  // group g base: 4 + g*44  -> conv_w, conv_b, then 2 blocks of 21 leaves
  // up base: 92 (c1_w,c1_b,c2_w,c2_b,c3_w,c3_b), last: 98,99
  const float* x_in = P(0);
  const float* cdp  = P(1);

  // ---- workspace layout (floats) ----
  float* base = (float*)d_ws;
  size_t off = 0;
  auto alloc = [&](size_t n) { float* p = base + off; off += n; return p; };
  float* t0 = alloc((size_t)Lc * DIMc);
  float* t1 = alloc((size_t)Lc * DIMc);
  float* xb = alloc((size_t)Lc * DIMc);
  float* tg = alloc((size_t)Lc * DIMc);
  float* xn = alloc((size_t)Lc * DIMc);
  float* md = alloc(128);
  float* scratch = base + off;                 // shared mamba / upsample region
  // mamba views
  float* xz  = scratch;                        // L*128
  float* xi  = xz  + (size_t)Lc * 128;         // L*64
  float* dbl = xi  + (size_t)Lc * DIc;         // L*34
  float* dlt = dbl + (size_t)Lc * 34;          // L*64
  float* ysc = dlt + (size_t)Lc * DIc;         // L*64
  float* yg  = ysc + (size_t)Lc * DIc;         // L*64
  float* hid = yg  + (size_t)Lc * DIc;         // L*64

  const size_t tbytes = (size_t)Lc * DIMc * sizeof(float);

  // ---- first conv: (1,1,F,H,W) -> channel-last (F,H,W,32) ----
  conv3d_cl<<<cdiv(Lc * DIMc, 256), 256, 0, stream>>>(
      x_in, P(2), P(3), nullptr, t0, Fc, Hc, Wc, 1, DIMc, 3, 3, 3, 1, 1, 1);
  hipMemcpyAsync(xb, t0, tbytes, hipMemcpyDeviceToDevice, stream);

  float* t = t0;
  float* tOther = t1;

  for (int g = 0; g < 2; ++g) {
    int gb = 4 + g * 44;
    hipMemcpyAsync(tg, t, tbytes, hipMemcpyDeviceToDevice, stream);
    for (int blk = 0; blk < 2; ++blk) {
      int q = gb + 2 + blk * 21;   // leaf base for this block
      // q+0 ln1_g, +1 ln1_b, +2 vdim1_w, +3 skip1,
      // +4 in_w, +5 mconv_w, +6 mconv_b, +7 x_proj_w, +8 dt_w, +9 dt_b,
      // +10 A_log, +11 D, +12 out_w,
      // +13 ln2_g, +14 ln2_b, +15 vdim2_w, +16 skip2,
      // +17 fc1_w, +18 fc1_b, +19 fc2_w, +20 fc2_b
      mod_k<<<1, 64, 0, stream>>>(cdp, P(q + 2), md, 128, 2 * DIMc);
      vdim_k<<<cdiv(Lc, 256), 256, 0, stream>>>(t, P(q), P(q + 1), md, xn, Lc, DIMc);
      // Mamba3D
      gemm<1, 8>(xn, P(q + 4), nullptr, nullptr, nullptr, xz, Lc, 2 * DIc, 0, stream);
      dwconv_silu_k<<<cdiv(Lc * DIc, 256), 256, 0, stream>>>(xz, P(q + 5), P(q + 6), xi, Lc);
      gemm<2, 3>(xi, P(q + 7), nullptr, nullptr, nullptr, dbl, Lc, 34, 0, stream);
      delta_k<<<cdiv(Lc * DIc, 256), 256, 0, stream>>>(dbl, P(q + 8), P(q + 9), dlt, Lc);
      scan_k<<<DIc / 16, 256, 0, stream>>>(dlt, dbl, xi, P(q + 10), ysc, Lc);
      gate_k<<<cdiv(Lc * DIc, 256), 256, 0, stream>>>(ysc, xi, xz, P(q + 11), yg, Lc);
      // t = t*skip1 + y @ out_w   (in-place residual epilogue)
      gemm<2, 2>(yg, P(q + 12), nullptr, t, P(q + 3), t, Lc, DIMc, 0, stream);
      // MLP branch
      mod_k<<<1, 64, 0, stream>>>(cdp, P(q + 15), md, 128, 2 * DIMc);
      vdim_k<<<cdiv(Lc, 256), 256, 0, stream>>>(t, P(q + 13), P(q + 14), md, xn, Lc, DIMc);
      gemm<1, 4>(xn, P(q + 17), P(q + 18), nullptr, nullptr, hid, Lc, 2 * DIMc, 2, stream);
      gemm<2, 2>(hid, P(q + 19), P(q + 20), t, P(q + 16), t, Lc, DIMc, 0, stream);
    }
    // group conv + residual (can't run in place: reads neighbors)
    conv3d_cl<<<cdiv(Lc * DIMc, 256), 256, 0, stream>>>(
        t, P(gb), P(gb + 1), tg, tOther, Fc, Hc, Wc, DIMc, DIMc, 3, 3, 3, 1, 1, 1);
    float* tmp = t; t = tOther; tOther = tmp;
  }

  // ---- reconstruction / upsampler ----
  // overlay upsample buffers on the (now dead) mamba scratch region
  float* v0 = scratch;                         // (F,H ,W,32)  L*32
  float* u1 = v0 + (size_t)Lc * DIMc;          // (F,H ,W,64)  L*64
  float* v1 = u1 + (size_t)Lc * 64;            // (F,2H,W,32)  L*64
  float* u2 = v1 + (size_t)Lc * 64;            // (F,2H,W,64)  L*128
  float* v2 = u2 + (size_t)Lc * 128;           // (F,4H,W,32)  L*128
  float* v3 = v2 + (size_t)Lc * 128;           // (F,4H,W,32)  L*128

  add_k<<<cdiv(Lc * DIMc, 256), 256, 0, stream>>>(t, xb, v0, Lc * DIMc);

  conv3d_cl<<<cdiv(Fc * Hc * Wc * 64, 256), 256, 0, stream>>>(
      v0, P(92), P(93), nullptr, u1, Fc, Hc, Wc, DIMc, 64, 1, 3, 3, 0, 1, 1);
  ps_leaky_k<<<cdiv(Fc * H2c * Wc * DIMc, 256), 256, 0, stream>>>(u1, v1, Fc, Hc, Wc, DIMc);

  conv3d_cl<<<cdiv(Fc * H2c * Wc * 64, 256), 256, 0, stream>>>(
      v1, P(94), P(95), nullptr, u2, Fc, H2c, Wc, DIMc, 64, 1, 3, 3, 0, 1, 1);
  ps_leaky_k<<<cdiv(Fc * H4c * Wc * DIMc, 256), 256, 0, stream>>>(u2, v2, Fc, H2c, Wc, DIMc);

  conv3d_cl<<<cdiv(Fc * H4c * Wc * DIMc, 256), 256, 0, stream>>>(
      v2, P(96), P(97), nullptr, v3, Fc, H4c, Wc, DIMc, DIMc, 1, 3, 3, 0, 1, 1);

  conv3d_cl<<<cdiv(Fc * H4c * Wc * 1, 256), 256, 0, stream>>>(
      v3, P(98), P(99), nullptr, (float*)d_out, Fc, H4c, Wc, DIMc, 1, 3, 3, 3, 1, 1, 1);
}
